// CrossScoreQwen3VLTextAttention_14078902796907
// MI455X (gfx1250) — matrix operand order repaired and checked by reading, not compiled, not run on hardware
//
#include <hip/hip_runtime.h>
#include <hip/hip_bf16.h>
#include <stdint.h>

// ---------------------------------------------------------------------------
// Types for CDNA5 WMMA (wave32, 16x16x32 bf16 -> f32)
// ---------------------------------------------------------------------------
typedef __bf16 bf16_t;
typedef __attribute__((ext_vector_type(16))) __bf16 v16bf;
typedef __attribute__((ext_vector_type(8)))  __bf16 v8bf;
typedef __attribute__((ext_vector_type(8)))  float  v8f;
typedef __attribute__((ext_vector_type(4)))  float  v4f;

#define S_LEN 2048
#define HID_  2048
#define NH_   16
#define NKV_  8
#define HD_   128
#define EPS_  1e-6f
#define SCALE_ 0.08838834764831845f   // 1/sqrt(128)

__device__ inline int lane_id() { return (int)__builtin_amdgcn_mbcnt_lo(~0u, 0u); }

// xor-shuffle via ds_bpermute (wave32-safe, no header wave-size assumptions)
__device__ inline float xor_shf(float v, int mask) {
  int src = (lane_id() ^ mask) << 2;
  int r = __builtin_amdgcn_ds_bpermute(src, __builtin_bit_cast(int, v));
  return __builtin_bit_cast(float, r);
}

__device__ inline v8f zero8() {
  v8f z;
#pragma unroll
  for (int i = 0; i < 8; ++i) z[i] = 0.0f;
  return z;
}

__device__ inline v8f wmma_bf16(v16bf a, v16bf b, v8f c) {
  // (neg_a, A, neg_b, B, c_mod, C, reuse_a, reuse_b)
  return __builtin_amdgcn_wmma_f32_16x16x32_bf16(false, a, false, b, (short)0, c,
                                                 false, false);
}

// A-fragment (16x32, 16-bit): lane holds K = {base..base+7, base+16..base+23}
__device__ inline v16bf load_a_frag(const bf16_t* base) {
  v8bf lo = *(const v8bf*)(base);
  v8bf hi = *(const v8bf*)(base + 16);
  v16bf r;
#pragma unroll
  for (int i = 0; i < 8; ++i) { r[i] = lo[i]; r[i + 8] = hi[i]; }
  return r;
}

// B-fragment (32x16, 16-bit): lane holds K = {base..base+7, base+16..base+23} of one column
__device__ inline v16bf load_b_frag(const bf16_t* base) {
  v8bf lo = *(const v8bf*)(base);
  v8bf hi = *(const v8bf*)(base + 16);
  v16bf r;
#pragma unroll
  for (int i = 0; i < 8; ++i) { r[i] = lo[i]; r[i + 8] = hi[i]; }
  return r;
}

// ---------------------------------------------------------------------------
// CDNA5 async global->LDS copy (GLOBAL_LOAD_ASYNC_TO_LDS_B128, ASYNCcnt).
// Per active lane: LDS[lds_addr .. +15] = MEM[gaddr .. +15], bypassing VGPRs.
// LDS byte address = low 32 bits of the generic shared pointer (aperture rule).
// ---------------------------------------------------------------------------
__device__ inline void async_copy16(void* lds_ptr, const void* gaddr) {
  unsigned lds_addr = (unsigned)(size_t)lds_ptr;
  asm volatile("global_load_async_to_lds_b128 %0, %1, off"
               :: "v"(lds_addr), "v"(gaddr)
               : "memory");
}

__device__ inline void wait_async() {
#if __has_builtin(__builtin_amdgcn_s_wait_asynccnt)
  __builtin_amdgcn_s_wait_asynccnt(0);
#else
  asm volatile("s_wait_asynccnt 0" ::: "memory");
#endif
}

// ---------------------------------------------------------------------------
// GEMM: C[MxN](f32) = A[MxK] * B[KxN](f32->bf16). A is f32 (converted) or bf16.
// 128x128 block tile, 8 waves of 32x64 (2x4 WMMA accumulators), K step 32.
// bf16-A path stages its tile with async-to-LDS DMA.
// ---------------------------------------------------------------------------
template <bool A_BF16>
__global__ __launch_bounds__(256) void gemm128_wmma(const void* __restrict__ Av,
                                                    const float* __restrict__ B,
                                                    float* __restrict__ C,
                                                    int N, int K) {
  __shared__ bf16_t ldsA[128 * 32];  // [row][k]
  __shared__ bf16_t ldsB[128 * 32];  // [n][k]  (K-major so frags are contiguous)

  const int tid  = threadIdx.x;
  const int lane = tid & 31;
  const int w    = tid >> 5;   // 0..7
  const int wr   = w >> 1;     // 0..3 : 32-row strip
  const int wc   = w & 1;      // 0..1 : 64-col strip
  const int l15  = lane & 15;
  const int hi   = lane >> 4;  // 0/1 half-wave
  const int brow = blockIdx.y * 128;
  const int bcol = blockIdx.x * 128;

  v8f acc[2][4];
#pragma unroll
  for (int m = 0; m < 2; ++m)
#pragma unroll
    for (int n = 0; n < 4; ++n) acc[m][n] = zero8();

  const int kTiles = K / 32;
  for (int kt = 0; kt < kTiles; ++kt) {
    const int k0 = kt * 32;
    __syncthreads();
    if (kt + 1 < kTiles) {  // emits global_prefetch_b8 for next B tile
      __builtin_prefetch((const void*)(B + (size_t)(k0 + 32) * N + bcol + lane * 16), 0, 1);
    }
    // ---- stage A (128x32) ----
    if (A_BF16) {
      const bf16_t* A = (const bf16_t*)Av;
#pragma unroll
      for (int j = 0; j < 2; ++j) {
        int idx = tid + 256 * j;            // 512 16B chunks
        int row = idx >> 2, c8 = idx & 3;
        async_copy16(&ldsA[row * 32 + c8 * 8],
                     A + (size_t)(brow + row) * K + k0 + c8 * 8);
      }
    } else {
      const float* A = (const float*)Av;
#pragma unroll
      for (int j = 0; j < 4; ++j) {
        int idx = tid + 256 * j;            // 1024 float4 chunks
        int row = idx >> 3, c4 = idx & 7;
        v4f v = *(const v4f*)(A + (size_t)(brow + row) * K + k0 + c4 * 4);
#pragma unroll
        for (int u = 0; u < 4; ++u) ldsA[row * 32 + c4 * 4 + u] = (bf16_t)v[u];
      }
    }
    // ---- stage B (32x128) transposed into [n][k] (needs f32->bf16 VALU) ----
#pragma unroll
    for (int j = 0; j < 4; ++j) {
      int idx = tid + 256 * j;              // 1024 float4 chunks
      int k = idx >> 5, n4 = idx & 31;
      v4f v = *(const v4f*)(B + (size_t)(k0 + k) * N + bcol + n4 * 4);
#pragma unroll
      for (int u = 0; u < 4; ++u) ldsB[(n4 * 4 + u) * 32 + k] = (bf16_t)v[u];
    }
    if (A_BF16) wait_async();  // drain this wave's ASYNCcnt before the barrier
    __syncthreads();
    // ---- fragments + 8 WMMAs ----
    v16bf af[2];
#pragma unroll
    for (int m = 0; m < 2; ++m) {
      int row = wr * 32 + m * 16 + l15;
      af[m] = load_a_frag(&ldsA[row * 32 + hi * 8]);
    }
#pragma unroll
    for (int n = 0; n < 4; ++n) {
      int col = wc * 64 + n * 16 + l15;
      v16bf bf = load_b_frag(&ldsB[col * 32 + hi * 8]);
#pragma unroll
      for (int m = 0; m < 2; ++m) acc[m][n] = wmma_bf16(af[m], bf, acc[m][n]);
    }
  }
  // ---- writeback (C layout: vgpr i -> row i (+8 for hi half), col = l15) ----
#pragma unroll
  for (int m = 0; m < 2; ++m)
#pragma unroll
    for (int n = 0; n < 4; ++n) {
      int col = bcol + wc * 64 + n * 16 + l15;
#pragma unroll
      for (int i = 0; i < 8; ++i) {
        int row = brow + wr * 32 + m * 16 + i + hi * 8;
        C[(size_t)row * N + col] = acc[m][n][i];
      }
    }
}

// ---------------------------------------------------------------------------
// RMS-norm + RoPE + repack to head-major bf16. One wave per (s, head) row.
// raw: (S, nheads*128) f32   out: (nheads, S, 128) bf16
// ---------------------------------------------------------------------------
__global__ __launch_bounds__(256) void rmsnorm_rope_pack(
    const float* __restrict__ raw, const float* __restrict__ normw,
    const float* __restrict__ cosp, const float* __restrict__ sinp,
    bf16_t* __restrict__ outp, int nheads, int do_norm_rope) {
  int wave = (int)((blockIdx.x * blockDim.x + threadIdx.x) >> 5);
  int lane = threadIdx.x & 31;
  int total = S_LEN * nheads;
  if (wave >= total) return;
  int s = wave / nheads;
  int h = wave - s * nheads;

  v4f xv = *(const v4f*)(raw + ((size_t)s * nheads + h) * HD_ + lane * 4);
  float x[4];
#pragma unroll
  for (int j = 0; j < 4; ++j) x[j] = xv[j];

  if (do_norm_rope) {
    float ss = x[0] * x[0] + x[1] * x[1] + x[2] * x[2] + x[3] * x[3];
#pragma unroll
    for (int m = 1; m < 32; m <<= 1) ss += xor_shf(ss, m);
    float inv = rsqrtf(ss * (1.0f / HD_) + EPS_);
#pragma unroll
    for (int j = 0; j < 4; ++j) x[j] = x[j] * inv * normw[lane * 4 + j];
    // RoPE: partner d^64 lives in lane^16 at the same sub-index
#pragma unroll
    for (int j = 0; j < 4; ++j) {
      int d = lane * 4 + j;
      float part = xor_shf(x[j], 16);
      float rot = (d < 64) ? -part : part;
      float c = cosp[(size_t)s * HD_ + d];
      float sn = sinp[(size_t)s * HD_ + d];
      x[j] = x[j] * c + rot * sn;
    }
  }
  bf16_t* dst = outp + ((size_t)h * S_LEN + s) * HD_ + lane * 4;
#pragma unroll
  for (int j = 0; j < 4; ++j) dst[j] = (bf16_t)x[j];
}

// ---------------------------------------------------------------------------
// Flash attention with WMMA. Block = 256 thr (8 waves), tile = 128 q x 128 kv.
// grid = (S/128, NH). Q/K/V head-major bf16. Writes attn_out (S, NH*128) bf16
// and accumulates cross-score sums (v2t at cross[h], t2v at cross[16+h]).
// Q and K tiles staged with async-to-LDS DMA; V needs a transpose (VALU path).
// ---------------------------------------------------------------------------
__global__ __launch_bounds__(256) void attn_flash_wmma(
    const bf16_t* __restrict__ qb, const bf16_t* __restrict__ kb,
    const bf16_t* __restrict__ vb, const unsigned char* __restrict__ vmask,
    bf16_t* __restrict__ attn_out, float* __restrict__ cross) {
  __shared__ bf16_t ldsKP[128 * 128];  // Q staging -> K tile -> P tile (aliased)
  __shared__ bf16_t ldsVt[128 * 128];  // V transposed: [d][key]

  const int tid = threadIdx.x, lane = tid & 31, w = tid >> 5;
  const int l15 = lane & 15, hi = lane >> 4;
  const int qt = blockIdx.x, h = blockIdx.y, kvh = h >> 1;  // N_REP = 2
  const bf16_t* qh = qb + (size_t)h * S_LEN * HD_;
  const bf16_t* kh = kb + (size_t)kvh * S_LEN * HD_;
  const bf16_t* vh = vb + (size_t)kvh * S_LEN * HD_;

  // ---- stage Q tile (async DMA), preload per-wave A fragments ----
#pragma unroll
  for (int j = 0; j < 8; ++j) {
    int idx = tid + 256 * j;
    int row = idx >> 4, c8 = idx & 15;
    async_copy16(&ldsKP[row * 128 + c8 * 8],
                 qh + (size_t)(qt * 128 + row) * HD_ + c8 * 8);
  }
  wait_async();
  __syncthreads();
  v16bf qf[4];
  {
    int row = w * 16 + l15;
#pragma unroll
    for (int kk = 0; kk < 4; ++kk)
      qf[kk] = load_a_frag(&ldsKP[row * 128 + kk * 32 + hi * 8]);
  }

  // per-lane-group row state (rows r = i + 8*hi of this wave's 16-row strip)
  float mrow[8], lrow[8], trow[8];
#pragma unroll
  for (int i = 0; i < 8; ++i) { mrow[i] = -1e30f; lrow[i] = 0.f; trow[i] = 0.f; }
  v8f oacc[8];
#pragma unroll
  for (int n = 0; n < 8; ++n) oacc[n] = zero8();

  for (int kt = 0; kt <= qt; ++kt) {
    __syncthreads();  // prior-iter P/V reads (and initial q-frag reads) done
    // ---- stage K tile [key][d] via async DMA ----
#pragma unroll
    for (int j = 0; j < 8; ++j) {
      int idx = tid + 256 * j;
      int row = idx >> 4, c8 = idx & 15;
      async_copy16(&ldsKP[row * 128 + c8 * 8],
                   kh + (size_t)(kt * 128 + row) * HD_ + c8 * 8);
    }
    // ---- stage V transposed [d][key] (VALU transpose path) ----
#pragma unroll
    for (int j = 0; j < 8; ++j) {
      int idx = tid + 256 * j;
      int key = idx >> 4, d8 = idx & 15;
      v8bf v = *(const v8bf*)(vh + (size_t)(kt * 128 + key) * HD_ + d8 * 8);
#pragma unroll
      for (int u = 0; u < 8; ++u) ldsVt[(d8 * 8 + u) * 128 + key] = v[u];
    }
    wait_async();  // drain this wave's async K-tile transfers
    __syncthreads();

    // ---- S = Q * K^T : 8 n-tiles x 4 k-steps ----
    v8f sacc[8];
#pragma unroll
    for (int t = 0; t < 8; ++t) sacc[t] = zero8();
#pragma unroll
    for (int kk = 0; kk < 4; ++kk)
#pragma unroll
      for (int t = 0; t < 8; ++t) {
        int key = t * 16 + l15;
        v16bf bf = load_b_frag(&ldsKP[key * 128 + kk * 32 + hi * 8]);
        sacc[t] = wmma_bf16(qf[kk], bf, sacc[t]);
      }
    __syncthreads();  // K tile fully consumed; ldsKP reusable for P

    // ---- scale + analytic causal mask + text flags per column ----
    float tf[8];
#pragma unroll
    for (int t = 0; t < 8; ++t) {
      int kcol = kt * 128 + t * 16 + l15;
      tf[t] = vmask[kcol] ? 0.0f : 1.0f;
    }
    const int rbase = w * 16 + hi * 8;  // row within q-tile for element i=0
#pragma unroll
    for (int t = 0; t < 8; ++t)
#pragma unroll
      for (int i = 0; i < 8; ++i) {
        float sv = sacc[t][i] * SCALE_;
        if (kt == qt) {
          int col = t * 16 + l15;
          if (col > rbase + i) sv = -1e30f;
        }
        sacc[t][i] = sv;
      }

    // ---- online softmax: row max over 128 cols (8 tiles x 16 lanes) ----
    float mloc[8];
#pragma unroll
    for (int i = 0; i < 8; ++i) {
      float mm = sacc[0][i];
#pragma unroll
      for (int t = 1; t < 8; ++t) mm = fmaxf(mm, sacc[t][i]);
#pragma unroll
      for (int msk = 1; msk < 16; msk <<= 1) mm = fmaxf(mm, xor_shf(mm, msk));
      mloc[i] = mm;
    }
    float rs[8], rst[8], scl[8];
#pragma unroll
    for (int i = 0; i < 8; ++i) {
      float mn = fmaxf(mrow[i], mloc[i]);
      scl[i] = __expf(mrow[i] - mn);
      mrow[i] = mn; rs[i] = 0.f; rst[i] = 0.f;
    }
    // ---- P = exp(S-m), accumulate row sums, spill P to LDS (wave-local) ----
#pragma unroll
    for (int t = 0; t < 8; ++t)
#pragma unroll
      for (int i = 0; i < 8; ++i) {
        float p = __expf(sacc[t][i] - mrow[i]);
        rs[i] += p;
        rst[i] += p * tf[t];
        ldsKP[(w * 16 + i + hi * 8) * 128 + t * 16 + l15] = (bf16_t)p;
      }
#pragma unroll
    for (int i = 0; i < 8; ++i) {
      float a = rs[i], b = rst[i];
#pragma unroll
      for (int msk = 1; msk < 16; msk <<= 1) {
        a += xor_shf(a, msk);
        b += xor_shf(b, msk);
      }
      lrow[i] = lrow[i] * scl[i] + a;
      trow[i] = trow[i] * scl[i] + b;
#pragma unroll
      for (int n = 0; n < 8; ++n) oacc[n][i] *= scl[i];
    }
    __syncthreads();  // P visible (also orders vs. compiler reordering)

    // ---- O += P * V : 4 k-steps x 8 d-tiles ----
#pragma unroll
    for (int st = 0; st < 4; ++st) {
      int arow = w * 16 + l15;
      v16bf pf = load_a_frag(&ldsKP[arow * 128 + st * 32 + hi * 8]);
#pragma unroll
      for (int n = 0; n < 8; ++n) {
        int dcol = n * 16 + l15;
        v16bf vf = load_b_frag(&ldsVt[dcol * 128 + st * 32 + hi * 8]);
        oacc[n] = wmma_bf16(pf, vf, oacc[n]);
      }
    }
  }

  // ---- epilogue: O/l -> bf16 attn_out, plus cross-score accumulation ----
#pragma unroll
  for (int n = 0; n < 8; ++n) {
    int d = n * 16 + l15;
#pragma unroll
    for (int i = 0; i < 8; ++i) {
      int row = qt * 128 + w * 16 + i + hi * 8;
      float o = oacc[n][i] / lrow[i];
      attn_out[(size_t)row * HID_ + h * HD_ + d] = (bf16_t)o;
    }
  }
  float v2t = 0.f, t2v = 0.f;
  if (l15 == 0) {  // row state replicated across the 16-lane group
#pragma unroll
    for (int i = 0; i < 8; ++i) {
      int row = qt * 128 + w * 16 + i + hi * 8;
      float frac = trow[i] / lrow[i];          // sum_{k in text} attn
      if (vmask[row]) v2t += frac;             // vision query -> text keys
      else            t2v += (1.0f - frac);    // text query -> vision keys
    }
  }
  v2t += xor_shf(v2t, 16);
  t2v += xor_shf(t2v, 16);
  if (lane == 0) {
    atomicAdd(&cross[h], v2t);
    atomicAdd(&cross[NH_ + h], t2v);
  }
}

// ---------------------------------------------------------------------------
__global__ void zero_cross(float* cross) {
  if (threadIdx.x < 2 * NH_) cross[threadIdx.x] = 0.0f;
}

__global__ __launch_bounds__(256) void finalize_cross(
    const float* __restrict__ cross, const unsigned char* __restrict__ vmask,
    float* __restrict__ out) {
  __shared__ float red[256];
  int tid = threadIdx.x;
  float s = 0.f;
  for (int i = tid; i < S_LEN; i += 256) s += vmask[i] ? 1.0f : 0.0f;
  red[tid] = s;
  __syncthreads();
  for (int off = 128; off > 0; off >>= 1) {
    if (tid < off) red[tid] += red[tid + off];
    __syncthreads();
  }
  float nv = red[0];
  float nt = (float)S_LEN - nv;
  if (tid < NH_) {
    float valid = (nv > 0.f && nt > 0.f) ? 1.0f : 0.0f;
    float v2t = cross[tid] / fmaxf(nv, 1.0f);
    float t2v = cross[NH_ + tid] / fmaxf(nt, 1.0f);
    out[(size_t)S_LEN * HID_ + tid] = valid * (v2t + t2v);
  }
}

// ---------------------------------------------------------------------------
extern "C" void kernel_launch(void* const* d_in, const int* in_sizes, int n_in,
                              void* d_out, int out_size, void* d_ws, size_t ws_size,
                              hipStream_t stream) {
  (void)in_sizes; (void)n_in; (void)out_size; (void)ws_size;
  const float* hs   = (const float*)d_in[0];
  const float* wq   = (const float*)d_in[1];
  const float* wk   = (const float*)d_in[2];
  const float* wv   = (const float*)d_in[3];
  const float* wo   = (const float*)d_in[4];
  const float* qnw  = (const float*)d_in[5];
  const float* knw  = (const float*)d_in[6];
  const float* cosp = (const float*)d_in[7];
  const float* sinp = (const float*)d_in[8];
  // d_in[9] attention_mask: causal, applied analytically (saves 16 MB traffic)
  const unsigned char* vmask = (const unsigned char*)d_in[10];
  // d_in[11] text_mask == !vision_mask, derived on the fly
  float* out = (float*)d_out;

  char* ws = (char*)d_ws;
  const size_t MB = (size_t)1024 * 1024;
  float*  qraw  = (float*)(ws + 0 * MB);    // 16 MB  (S x 2048 f32)
  float*  kraw  = (float*)(ws + 16 * MB);   //  8 MB  (S x 1024 f32)
  float*  vraw  = (float*)(ws + 24 * MB);   //  8 MB
  bf16_t* qbf   = (bf16_t*)(ws + 32 * MB);  //  8 MB  (NH, S, 128 bf16)
  bf16_t* kbf   = (bf16_t*)(ws + 40 * MB);  //  4 MB  (NKV, S, 128)
  bf16_t* vbf   = (bf16_t*)(ws + 44 * MB);  //  4 MB
  bf16_t* abf   = (bf16_t*)(ws + 48 * MB);  //  8 MB  (S, 2048 bf16)
  float*  cross = (float*)(ws + 56 * MB);   // 128 B

  dim3 blk(256);
  hipLaunchKernelGGL(zero_cross, dim3(1), dim3(32), 0, stream, cross);

  // QKV projections (bf16 WMMA, f32 accumulate)
  hipLaunchKernelGGL((gemm128_wmma<false>), dim3(16, 16), blk, 0, stream,
                     (const void*)hs, wq, qraw, 2048, 2048);
  hipLaunchKernelGGL((gemm128_wmma<false>), dim3(8, 16), blk, 0, stream,
                     (const void*)hs, wk, kraw, 1024, 2048);
  hipLaunchKernelGGL((gemm128_wmma<false>), dim3(8, 16), blk, 0, stream,
                     (const void*)hs, wv, vraw, 1024, 2048);

  // RMS-norm + RoPE + repack (one wave per row; 8 waves/block)
  hipLaunchKernelGGL(rmsnorm_rope_pack, dim3((S_LEN * NH_) / 8), blk, 0, stream,
                     qraw, qnw, cosp, sinp, qbf, NH_, 1);
  hipLaunchKernelGGL(rmsnorm_rope_pack, dim3((S_LEN * NKV_) / 8), blk, 0, stream,
                     kraw, knw, cosp, sinp, kbf, NKV_, 1);
  hipLaunchKernelGGL(rmsnorm_rope_pack, dim3((S_LEN * NKV_) / 8), blk, 0, stream,
                     vraw, qnw, cosp, sinp, vbf, NKV_, 0);

  // Fused causal flash attention + cross-score accumulation
  hipLaunchKernelGGL(attn_flash_wmma, dim3(S_LEN / 128, NH_), blk, 0, stream,
                     qbf, kbf, vbf, vmask, abf, cross);

  // Output projection straight into d_out
  hipLaunchKernelGGL((gemm128_wmma<true>), dim3(16, 16), blk, 0, stream,
                     (const void*)abf, wo, out, 2048, 2048);

  hipLaunchKernelGGL(finalize_cross, dim3(1), blk, 0, stream, cross, vmask, out);
}
